// TokenEncoder_14886356648631
// MI455X (gfx1250) — matrix operand order repaired and checked by reading, not compiled, not run on hardware
//
#include <hip/hip_runtime.h>
#include <hip/hip_bf16.h>

typedef __attribute__((ext_vector_type(16))) __bf16 v16bf;
typedef __attribute__((ext_vector_type(8)))  float  v8f;

#define Bsz   32
#define Lseq  512
#define Dh    1024
#define K2    2048   // 2*Dh (concat [x, h])
#define N4    4096   // 4*Dh (i,g,f,o gates)

struct U128x2 { uint4 lo, hi; };

// ---------------------------------------------------------------------------
// One-time: embedding gather + f32 -> bf16, layout e2[t][b][d] so each step
// reads a contiguous 64KB block.
// ---------------------------------------------------------------------------
__global__ __launch_bounds__(256)
void embed_bf16_kernel(const int* __restrict__ x, const float* __restrict__ embed,
                       __bf16* __restrict__ e2) {
    size_t i = (size_t)blockIdx.x * 256 + threadIdx.x;   // < L*B*D
    int d = (int)(i & (Dh - 1));
    int b = (int)((i >> 10) & (Bsz - 1));
    int t = (int)(i >> 15);
    int tok = x[b * Lseq + t];
    e2[i] = (__bf16)embed[(size_t)tok * Dh + d];
}

// ---------------------------------------------------------------------------
// One-time: pack W [K2 x N4] f32 row-major into per-lane WMMA B-tile layout:
// tile (nb, kb) covers K=kb*32..+31, N=nb*16..+15.  Within a tile, lane l<16
// holds (K=h, N=l) for h=0..15; lane l>=16 holds (K=16+h, N=l-16).
// packed index = ((nb*64 + kb)*32 + lane)*16 + h  -> lane reads 32B contiguous.
// ---------------------------------------------------------------------------
__global__ __launch_bounds__(256)
void packw_bf16_kernel(const float* __restrict__ W, __bf16* __restrict__ Wp) {
    size_t i = (size_t)blockIdx.x * 256 + threadIdx.x;   // < K2*N4
    int h    = (int)(i & 15);
    int lane = (int)((i >> 4) & 31);
    int kb   = (int)((i >> 9) & 63);
    int nb   = (int)(i >> 15);                           // 0..255
    int nn   = lane & 15;
    int kk   = (lane < 16) ? h : (16 + h);
    Wp[i] = (__bf16)W[(size_t)(kb * 32 + kk) * N4 + nb * 16 + nn];
}

// ---------------------------------------------------------------------------
// One-time: zero h ping-pong buffers (bf16) and c buffers (f32), both dirs.
// ---------------------------------------------------------------------------
__global__ __launch_bounds__(256)
void zero_state_kernel(__bf16* __restrict__ hF, __bf16* __restrict__ hB,
                       float* __restrict__ cF, float* __restrict__ cB) {
    int i = blockIdx.x * 256 + threadIdx.x;              // grid covers 65536
    if (i < 2 * Bsz * Dh) { hF[i] = (__bf16)0.f; hB[i] = (__bf16)0.f; }
    if (i < Bsz * Dh)     { cF[i] = 0.f;         cB[i] = 0.f; }
}

__device__ __forceinline__ float sigmoid_f(float x) {
    return 1.f / (1.f + __expf(-x));
}
__device__ __forceinline__ float tanh_f(float x) {
    return 1.f - 2.f / (__expf(2.f * x) + 1.f);
}

// ---------------------------------------------------------------------------
// One fused LSTM step.  Grid: 64 WGs x 128 threads (4 waves).
// WG handles d-slice [d0, d0+16); wave g computes gate g's 32x16 tile of
//   gates = [e_t | h_prev] @ W  via v_wmma_f32_16x16x32_bf16 (K2/32 = 64 iters,
//   2 M-tiles each).  Then gates exchanged through LDS, cell update done
//   in-kernel, h written (f32 to out, bf16 to hout for next step).
// dir=1 (backward over flipped seq) applies ResetCore masking to h and c.
// ---------------------------------------------------------------------------
__global__ __launch_bounds__(128)
void lstm_step_kernel(const __bf16* __restrict__ e2, const __bf16* __restrict__ Wp,
                      const float* __restrict__ bias, const int* __restrict__ lengths,
                      const __bf16* __restrict__ hin, __bf16* __restrict__ hout,
                      float* __restrict__ cbuf, float* __restrict__ out,
                      int t, int dir) {
    __shared__ float gl[4][Bsz][16];

    const int tid  = threadIdx.x;
    const int lane = tid & 31;
    const int gate = tid >> 5;              // 0..3 -> i,g,f,o
    const int d0   = blockIdx.x * 16;

    const int mrow = lane & 15;
    const int hi   = lane >> 4;             // lane group
    const int koff = hi * 16;               // A-layout K sub-offset
    const int row0 = mrow;                  // M-tile 0 rows (batch 0..15)
    const int row1 = 16 + mrow;             // M-tile 1 rows (batch 16..31)

    // ResetCore flags for backward pass (reset state where mask true)
    bool rz0 = false, rz1 = false;
    if (dir) {
        int pos = (Lseq - 1) - t;
        rz0 = pos < (lengths[row0] - 1);
        rz1 = pos < (lengths[row1] - 1);
    }

    const __bf16* eB = e2 + (size_t)t * Bsz * Dh;                     // [b][d]
    const int nb = gate * (Dh / 16) + (d0 >> 4);
    const __bf16* wB = Wp + (size_t)nb * 64 * 512 + (size_t)lane * 16;

    v8f acc0 = {}; v8f acc1 = {};

    // --- K phase 1: embedding half (K = 0..1023) ---
    for (int ki = 0; ki < 32; ++ki) {
        const int k0 = ki * 32;
        U128x2 ta, tb, tw;
        const __bf16* p0 = eB + (size_t)row0 * Dh + k0 + koff;
        ta.lo = *(const uint4*)(p0);
        ta.hi = *(const uint4*)(p0 + 8);
        const __bf16* p1 = eB + (size_t)row1 * Dh + k0 + koff;
        tb.lo = *(const uint4*)(p1);
        tb.hi = *(const uint4*)(p1 + 8);
        const __bf16* pw = wB + (size_t)ki * 512;
        tw.lo = *(const uint4*)(pw);
        tw.hi = *(const uint4*)(pw + 8);
        v16bf a0 = __builtin_bit_cast(v16bf, ta);
        v16bf a1 = __builtin_bit_cast(v16bf, tb);
        v16bf bt = __builtin_bit_cast(v16bf, tw);
        acc0 = __builtin_amdgcn_wmma_f32_16x16x32_bf16(false, a0, false, bt, (short)0, acc0, false, false);
        acc1 = __builtin_amdgcn_wmma_f32_16x16x32_bf16(false, a1, false, bt, (short)0, acc1, false, false);
    }
    // --- K phase 2: hidden-state half (K = 1024..2047), with reset mask ---
    for (int ki = 32; ki < 64; ++ki) {
        const int k0 = (ki - 32) * 32;
        U128x2 ta, tb, tw;
        const __bf16* p0 = hin + (size_t)row0 * Dh + k0 + koff;
        ta.lo = *(const uint4*)(p0);
        ta.hi = *(const uint4*)(p0 + 8);
        const __bf16* p1 = hin + (size_t)row1 * Dh + k0 + koff;
        tb.lo = *(const uint4*)(p1);
        tb.hi = *(const uint4*)(p1 + 8);
        if (rz0) { ta.lo = make_uint4(0,0,0,0); ta.hi = make_uint4(0,0,0,0); }
        if (rz1) { tb.lo = make_uint4(0,0,0,0); tb.hi = make_uint4(0,0,0,0); }
        const __bf16* pw = wB + (size_t)ki * 512;
        tw.lo = *(const uint4*)(pw);
        tw.hi = *(const uint4*)(pw + 8);
        v16bf a0 = __builtin_bit_cast(v16bf, ta);
        v16bf a1 = __builtin_bit_cast(v16bf, tb);
        v16bf bt = __builtin_bit_cast(v16bf, tw);
        acc0 = __builtin_amdgcn_wmma_f32_16x16x32_bf16(false, a0, false, bt, (short)0, acc0, false, false);
        acc1 = __builtin_amdgcn_wmma_f32_16x16x32_bf16(false, a1, false, bt, (short)0, acc1, false, false);
    }

    // C-layout: VGPR r -> (lanes 0-15: M=r, N=lane) (lanes 16-31: M=8+r, N=lane-16)
    const int col = mrow;
    #pragma unroll
    for (int r = 0; r < 8; ++r) {
        int rb = hi ? (r + 8) : r;
        gl[gate][rb][col]      = acc0[r];
        gl[gate][rb + 16][col] = acc1[r];
    }
    __syncthreads();

    // Cell update: 512 elements / 128 threads
    #pragma unroll
    for (int p = 0; p < 4; ++p) {
        int e = tid + p * 128;
        int b = e >> 4;
        int j = e & 15;
        int d = d0 + j;
        float gi = gl[0][b][j] + bias[d];
        float gg = gl[1][b][j] + bias[Dh + d];
        float gf = gl[2][b][j] + bias[2 * Dh + d];
        float go = gl[3][b][j] + bias[3 * Dh + d];
        float cp = cbuf[b * Dh + d];
        if (dir) {
            int pos = (Lseq - 1) - t;
            if (pos < (lengths[b] - 1)) cp = 0.f;
        }
        float fs = sigmoid_f(gf + 1.f);
        float is = sigmoid_f(gi);
        float gt = tanh_f(gg);
        float os = sigmoid_f(go);
        float cn = fs * cp + is * gt;
        float hn = os * tanh_f(cn);
        cbuf[b * Dh + d] = cn;
        hout[b * Dh + d] = (__bf16)hn;
        size_t orow = dir ? ((size_t)b * Lseq + (Lseq - 1 - t))
                          : ((size_t)b * Lseq + t);
        out[orow * (2 * Dh) + dir * Dh + d] = hn;
    }
}

// ---------------------------------------------------------------------------
extern "C" void kernel_launch(void* const* d_in, const int* in_sizes, int n_in,
                              void* d_out, int out_size, void* d_ws, size_t ws_size,
                              hipStream_t stream) {
    (void)in_sizes; (void)n_in; (void)out_size; (void)ws_size;
    const int*   x       = (const int*)d_in[0];
    const int*   lengths = (const int*)d_in[1];
    const float* embed   = (const float*)d_in[2];
    const float* W_fwd   = (const float*)d_in[3];
    const float* b_fwd   = (const float*)d_in[4];
    const float* W_bwd   = (const float*)d_in[5];
    const float* b_bwd   = (const float*)d_in[6];
    float* out = (float*)d_out;

    char* ws = (char*)d_ws;
    __bf16* e2  = (__bf16*)(ws);                       // L*B*D bf16   = 32 MB
    __bf16* Wfp = (__bf16*)(ws + 33554432ull);         // K2*N4 bf16   = 16 MB
    __bf16* Wbp = (__bf16*)(ws + 50331648ull);         // K2*N4 bf16   = 16 MB
    __bf16* hF  = (__bf16*)(ws + 67108864ull);         // 2*B*D bf16   = 128 KB
    __bf16* hB  = (__bf16*)(ws + 67239936ull);         // 2*B*D bf16   = 128 KB
    float*  cF  = (float*) (ws + 67371008ull);         // B*D f32      = 128 KB
    float*  cB  = (float*) (ws + 67502080ull);         // B*D f32      = 128 KB

    embed_bf16_kernel<<<65536, 256, 0, stream>>>(x, embed, e2);
    packw_bf16_kernel<<<32768, 256, 0, stream>>>(W_fwd, Wfp);
    packw_bf16_kernel<<<32768, 256, 0, stream>>>(W_bwd, Wbp);
    zero_state_kernel<<<256, 256, 0, stream>>>(hF, hB, cF, cB);

    // Forward recurrence: h ping-pong across steps (kernel boundary = sync)
    for (int t = 0; t < Lseq; ++t) {
        __bf16* hin  = hF + (size_t)(t & 1) * (Bsz * Dh);
        __bf16* hout = hF + (size_t)((t + 1) & 1) * (Bsz * Dh);
        lstm_step_kernel<<<64, 128, 0, stream>>>(e2, Wfp, b_fwd, lengths,
                                                 hin, hout, cF, out, t, 0);
    }
    // Backward recurrence (over flipped sequence, ResetCore masking)
    for (int t = 0; t < Lseq; ++t) {
        __bf16* hin  = hB + (size_t)(t & 1) * (Bsz * Dh);
        __bf16* hout = hB + (size_t)((t + 1) & 1) * (Bsz * Dh);
        lstm_step_kernel<<<64, 128, 0, stream>>>(e2, Wbp, b_bwd, lengths,
                                                 hin, hout, cB, out, t, 1);
    }
}
